// AHardPair_55748675502676
// MI455X (gfx1250) — matrix operand order repaired and checked by reading, not compiled, not run on hardware
//
#include <hip/hip_runtime.h>
#include <hip/hip_bf16.h>
#include <math.h>

typedef __attribute__((ext_vector_type(16))) __bf16 bf16x16;
typedef __attribute__((ext_vector_type(8)))  float  floatx8;

#define N_ROWS 4096
#define DIM    1024
#define TILE   128
#define KC     32
#define NBJ    (N_ROWS / TILE)   // 32 j-blocks
#define ALPHA_C 40.0f
#define BETA_C  20.0f

// ---------------------------------------------------------------------------
// Async copy: 16B per lane, global -> LDS, bypassing VGPRs (ASYNCcnt path).
// GVS mode: mem_addr = SGPR_base + sext(voff) ; LDS dest = per-lane lds offset.
// ---------------------------------------------------------------------------
__device__ __forceinline__ void async_b128(uint32_t lds, uint32_t voff, const void* sbase) {
    asm volatile("global_load_async_to_lds_b128 %0, %1, %2"
                 :: "v"(lds), "v"(voff), "s"(sbase)
                 : "memory");
}

// ---------------------------------------------------------------------------
// Kernel 1: split fp32 -> bf16 hi/lo and compute exact row squared norms.
// ---------------------------------------------------------------------------
__global__ __launch_bounds__(256)
void convert_sq_kernel(const float* __restrict__ x,
                       __bf16* __restrict__ xh, __bf16* __restrict__ xl,
                       float* __restrict__ sq) {
    const int row = blockIdx.x;
    const int t   = threadIdx.x;
    const float* xr = x + (size_t)row * DIM;
    float s = 0.0f;
    for (int c = t; c < DIM; c += 256) {
        float v = xr[c];
        s += v * v;
        __bf16 h  = (__bf16)v;
        float  hf = (float)h;
        __bf16 l  = (__bf16)(v - hf);
        xh[(size_t)row * DIM + c] = h;
        xl[(size_t)row * DIM + c] = l;
    }
    __shared__ float red[256];
    red[t] = s;
    __syncthreads();
    for (int off = 128; off > 0; off >>= 1) {
        if (t < off) red[t] += red[t + off];
        __syncthreads();
    }
    if (t == 0) sq[row] = red[0];
}

// ---------------------------------------------------------------------------
// Kernel 2: fused 128x128 G-tile via bf16 WMMA (hi/lo split, f32 accum),
// double-buffered LDS staged with GLOBAL_LOAD_ASYNC_TO_LDS_B128, plus the
// dist / masked-sum epilogue. Deterministic per-(row, j-block) partials.
// part layout: part[q * N_ROWS * NBJ + row * NBJ + bj], q in 0..7 =
//   {pos_logit, neg_logit, pos_exp_beta, neg_exp_beta, pos_d, neg_d, cnt_p, cnt_n}
// ---------------------------------------------------------------------------
__global__ __launch_bounds__(256)
void tile_kernel(const __bf16* __restrict__ xh, const __bf16* __restrict__ xl,
                 const float* __restrict__ sq, const int* __restrict__ tgt,
                 float* __restrict__ part) {
    const int bj = blockIdx.x;
    const int bi = blockIdx.y;
    const int t    = threadIdx.x;
    const int lane = t & 31;
    const int wave = t >> 5;
    const int lr   = lane & 15;   // N (col) index / M row index within 16
    const int hb   = lane >> 4;   // half-wave selector

    // Double-buffered staging tiles: 2 x (4 x 8KB) = 64 KB of the 320 KB WGP LDS.
    __shared__ __align__(16) __bf16 AhB[2][TILE * KC], AlB[2][TILE * KC];
    __shared__ __align__(16) __bf16 BhB[2][TILE * KC], BlB[2][TILE * KC];
    __shared__ float sqi[TILE], sqj[TILE];
    __shared__ int   ti[TILE],  tj[TILE];

    if (t < TILE) {
        sqi[t] = sq[bi * TILE + t];
        ti[t]  = tgt[bi * TILE + t];
    } else {
        int u = t - TILE;
        sqj[u] = sq[bj * TILE + u];
        tj[u]  = tgt[bj * TILE + u];
    }

    // Per-thread copy slots: 2 x 16B per tile array per chunk (512 uint4/tile).
    uint32_t dstOff[2], srcAOff[2], srcBOff[2];
#pragma unroll
    for (int p = 0; p < 2; ++p) {
        const int u = t + p * 256;
        const int r = u >> 2;
        const int q = u & 3;
        dstOff[p]  = (uint32_t)(r * KC + q * 8) * 2u;                      // bytes in tile
        srcAOff[p] = (uint32_t)((bi * TILE + r) * DIM + q * 8) * 2u;       // bytes in X
        srcBOff[p] = (uint32_t)((bj * TILE + r) * DIM + q * 8) * 2u;
    }
    // LDS byte offsets of the staging buffers (addr[31:0] of a shared pointer).
    const uint32_t ldsAh[2] = { (uint32_t)(uintptr_t)&AhB[0][0], (uint32_t)(uintptr_t)&AhB[1][0] };
    const uint32_t ldsAl[2] = { (uint32_t)(uintptr_t)&AlB[0][0], (uint32_t)(uintptr_t)&AlB[1][0] };
    const uint32_t ldsBh[2] = { (uint32_t)(uintptr_t)&BhB[0][0], (uint32_t)(uintptr_t)&BhB[1][0] };
    const uint32_t ldsBl[2] = { (uint32_t)(uintptr_t)&BlB[0][0], (uint32_t)(uintptr_t)&BlB[1][0] };

    auto issue = [&](int bufi, int kc) {
        const uint32_t kb = (uint32_t)kc * 2u;   // byte offset of K-chunk
#pragma unroll
        for (int p = 0; p < 2; ++p) {
            async_b128(ldsAh[bufi] + dstOff[p], srcAOff[p] + kb, xh);
            async_b128(ldsAl[bufi] + dstOff[p], srcAOff[p] + kb, xl);
            async_b128(ldsBh[bufi] + dstOff[p], srcBOff[p] + kb, xh);
            async_b128(ldsBl[bufi] + dstOff[p], srcBOff[p] + kb, xl);
        }
    };

    floatx8 acc[8];
#pragma unroll
    for (int i = 0; i < 8; ++i) acc[i] = (floatx8)0.0f;

    auto compute = [&](int bufi) {
        // A fragment (16x32 bf16): lane lr holds row M=lr; half hb holds
        // K = hb*8..hb*8+7 and K = 16+hb*8..+7 (ISA 16-bit A layout).
        union FragU { uint4 u[2]; bf16x16 v; } fah, fal;
        const int ar = wave * 16 + lr;
        const __bf16* Ah = &AhB[bufi][0];
        const __bf16* Al = &AlB[bufi][0];
        const __bf16* Bh = &BhB[bufi][0];
        const __bf16* Bl = &BlB[bufi][0];
        fah.u[0] = *(const uint4*)(Ah + ar * KC + hb * 8);
        fah.u[1] = *(const uint4*)(Ah + ar * KC + 16 + hb * 8);
        fal.u[0] = *(const uint4*)(Al + ar * KC + hb * 8);
        fal.u[1] = *(const uint4*)(Al + ar * KC + 16 + hb * 8);
#pragma unroll
        for (int tn = 0; tn < 8; ++tn) {
            // B fragment (32x16 bf16): lane lr = column N; half hb holds
            // K = hb*8..hb*8+7 and K = 16+hb*8..+7 (same as A).
            union FragU fbh, fbl;
            const int br = tn * 16 + lr;
            fbh.u[0] = *(const uint4*)(Bh + br * KC + hb * 8);
            fbh.u[1] = *(const uint4*)(Bh + br * KC + 16 + hb * 8);
            fbl.u[0] = *(const uint4*)(Bl + br * KC + hb * 8);
            fbl.u[1] = *(const uint4*)(Bl + br * KC + 16 + hb * 8);
            acc[tn] = __builtin_amdgcn_wmma_f32_16x16x32_bf16(
                false, fah.v, false, fbh.v, (short)0, acc[tn], false, false);
            acc[tn] = __builtin_amdgcn_wmma_f32_16x16x32_bf16(
                false, fah.v, false, fbl.v, (short)0, acc[tn], false, false);
            acc[tn] = __builtin_amdgcn_wmma_f32_16x16x32_bf16(
                false, fal.v, false, fbh.v, (short)0, acc[tn], false, false);
        }
    };

    // Software pipeline: chunk k+1 streams into buf^1 via the async unit while
    // WMMAs consume buf. Each chunk = 8 async B128s per thread; async loads
    // complete in order, so asynccnt<=8 == "previous chunk has landed".
    issue(0, 0);
    int buf = 0;
    for (int kc = 0; kc < DIM; kc += KC) {
        if (kc + KC < DIM) {
            issue(buf ^ 1, kc + KC);
            asm volatile("s_wait_asynccnt 0x8" ::: "memory");
        } else {
            asm volatile("s_wait_asynccnt 0x0" ::: "memory");
        }
        __syncthreads();
        compute(buf);
        __syncthreads();
        buf ^= 1;
    }

    // Epilogue: C layout — VGPR r holds M = r (lanes 0-15) / r+8 (lanes 16-31),
    // N = lane & 15. Fuse dist + masked sums, reduce over the 16 lanes that
    // share a row, write deterministic partials.
#pragma unroll
    for (int r = 0; r < 8; ++r) {
        const int il = wave * 16 + hb * 8 + r;    // local row in tile
        const int gi = bi * TILE + il;            // global row
        const float sqa = sqi[il];
        const int   ta  = ti[il];
        float sP = 0.f, sN = 0.f, sPL = 0.f, sNL = 0.f;
        float sPD = 0.f, sND = 0.f, sCP = 0.f, sCN = 0.f;
#pragma unroll
        for (int tn = 0; tn < 8; ++tn) {
            const int jl = tn * 16 + lr;
            const int gj = bj * TILE + jl;
            float g  = acc[tn][r];
            float d2 = sqa + sqj[jl] - 2.0f * g;
            float d  = sqrtf(fmaxf(d2, 1e-12f));
            bool same = (ta == tj[jl]);
            bool pos  = same && (gi != gj);
            bool neg  = !same;
            float ea = __expf(ALPHA_C * (1.0f - d));
            if (pos) {
                sP  += ea;
                sPL += __expf(BETA_C * (d - 0.8f));
                sPD += d;
                sCP += 1.0f;
            }
            if (neg) {
                sN  += ea;
                sNL += __expf(BETA_C * (1.1f - d));
                sND += d;
                sCN += 1.0f;
            }
        }
#pragma unroll
        for (int m = 8; m >= 1; m >>= 1) {
            sP  += __shfl_xor(sP,  m); sN  += __shfl_xor(sN,  m);
            sPL += __shfl_xor(sPL, m); sNL += __shfl_xor(sNL, m);
            sPD += __shfl_xor(sPD, m); sND += __shfl_xor(sND, m);
            sCP += __shfl_xor(sCP, m); sCN += __shfl_xor(sCN, m);
        }
        if (lr == 0) {
            const size_t base = (size_t)gi * NBJ + bj;
            const size_t stride = (size_t)N_ROWS * NBJ;
            part[0 * stride + base] = sP;
            part[1 * stride + base] = sN;
            part[2 * stride + base] = sPL;
            part[3 * stride + base] = sNL;
            part[4 * stride + base] = sPD;
            part[5 * stride + base] = sND;
            part[6 * stride + base] = sCP;
            part[7 * stride + base] = sCN;
        }
    }
}

// ---------------------------------------------------------------------------
// Kernel 3: per-row finalize (deterministic sum over 32 j-block partials).
// rowv layout: rowv[q * N_ROWS + row], q = {row_loss, PD, ND, CP, CN}
// ---------------------------------------------------------------------------
__global__ __launch_bounds__(256)
void rowfinal_kernel(const float* __restrict__ part, float* __restrict__ rowv) {
    const int row = blockIdx.x * blockDim.x + threadIdx.x;
    if (row >= N_ROWS) return;
    const size_t stride = (size_t)N_ROWS * NBJ;
    float q[8];
#pragma unroll
    for (int a = 0; a < 8; ++a) {
        float s = 0.f;
        const float* p = part + a * stride + (size_t)row * NBJ;
#pragma unroll
        for (int b = 0; b < NBJ; ++b) s += p[b];
        q[a] = s;
    }
    const float P = q[0], Ng = q[1], PL = q[2], NL = q[3];
    const float a_lr = 1.0f - P / (P + Ng);
    rowv[0 * N_ROWS + row] = a_lr * (logf(PL) + logf(NL));
    rowv[1 * N_ROWS + row] = q[4];
    rowv[2 * N_ROWS + row] = q[5];
    rowv[3 * N_ROWS + row] = q[6];
    rowv[4 * N_ROWS + row] = q[7];
}

// ---------------------------------------------------------------------------
// Kernel 4: single-block reduce to the 4 outputs.
// ---------------------------------------------------------------------------
__global__ __launch_bounds__(256)
void out_kernel(const float* __restrict__ rowv, float* __restrict__ out) {
    __shared__ float red[5][256];
    const int t = threadIdx.x;
    float s[5] = {0.f, 0.f, 0.f, 0.f, 0.f};
    for (int r = t; r < N_ROWS; r += 256) {
#pragma unroll
        for (int a = 0; a < 5; ++a) s[a] += rowv[a * N_ROWS + r];
    }
#pragma unroll
    for (int a = 0; a < 5; ++a) red[a][t] = s[a];
    __syncthreads();
    for (int off = 128; off > 0; off >>= 1) {
        if (t < off) {
#pragma unroll
            for (int a = 0; a < 5; ++a) red[a][t] += red[a][t + off];
        }
        __syncthreads();
    }
    if (t == 0) {
        out[0] = red[0][0] / (float)N_ROWS;   // loss
        out[1] = 0.0f;                        // accuracy (always 0 in ref)
        out[2] = red[1][0] / red[3][0];       // pos_d
        out[3] = red[2][0] / red[4][0];       // neg_d
    }
}

// ---------------------------------------------------------------------------
extern "C" void kernel_launch(void* const* d_in, const int* in_sizes, int n_in,
                              void* d_out, int out_size, void* d_ws, size_t ws_size,
                              hipStream_t stream) {
    (void)in_sizes; (void)n_in; (void)out_size; (void)ws_size;
    const float* x   = (const float*)d_in[0];
    const int*   tgt = (const int*)d_in[1];
    float*       out = (float*)d_out;

    // Workspace layout (all 16B aligned):
    //   sq   : N_ROWS floats                    (16 KB)
    //   part : 8 * N_ROWS * NBJ floats          ( 4 MB)
    //   rowv : 5 * N_ROWS floats                (80 KB)
    //   xh   : N_ROWS * DIM bf16                ( 8 MB)
    //   xl   : N_ROWS * DIM bf16                ( 8 MB)
    float* sq   = (float*)d_ws;
    float* part = sq + N_ROWS;
    float* rowv = part + (size_t)8 * N_ROWS * NBJ;
    __bf16* xh  = (__bf16*)(rowv + 5 * N_ROWS);
    __bf16* xl  = xh + (size_t)N_ROWS * DIM;

    convert_sq_kernel<<<N_ROWS, 256, 0, stream>>>(x, xh, xl, sq);

    dim3 grid(NBJ, N_ROWS / TILE);   // (bj, bi) = 32 x 32 tiles
    tile_kernel<<<grid, 256, 0, stream>>>(xh, xl, sq, tgt, part);

    rowfinal_kernel<<<N_ROWS / 256, 256, 0, stream>>>(part, rowv);
    out_kernel<<<1, 256, 0, stream>>>(rowv, out);
}
